// StatefulMemoryModule_56367150793444
// MI455X (gfx1250) — matrix operand order repaired and checked by reading, not compiled, not run on hardware
//
#include <hip/hip_runtime.h>
#include <hip/hip_bf16.h>

typedef _Float16 v16h __attribute__((ext_vector_type(16)));
typedef _Float16 v8h  __attribute__((ext_vector_type(8)));
typedef float    v8f  __attribute__((ext_vector_type(8)));

#define B_DIM 512
#define M_DIM 512
#define D_DIM 128
#define NTILES (M_DIM / 16)

__device__ __forceinline__ float fast_rcp(float x) {
    return __builtin_amdgcn_rcpf(x);   // raw v_rcp_f32, no IEEE-div expansion
}
__device__ __forceinline__ float fast_sigmoid(float x) {
    return fast_rcp(1.0f + __expf(-x));
}
__device__ __forceinline__ float fast_tanh(float x) {
    float e = __expf(2.0f * x);
    return (e - 1.0f) * fast_rcp(e + 1.0f);
}

// Load one 16x32 f16 A-fragment (CDNA5 wave32 layout) from a row-major
// [16][128] f16 LDS tile. Per lane: row = lane&15; the 16 halfs split into
// two contiguous 16B chunks -> two ds_load_b128:
//   elems 0..7  -> k = kbase + (hi?8:0) + 0..7
//   elems 8..15 -> k = kbase + (hi?8:0) + 16..23
__device__ __forceinline__ v16h load_Afrag(const _Float16* lds, int row, int kbase, int hi) {
    const _Float16* base = lds + row * D_DIM + kbase + (hi ? 8 : 0);
    v8h lo = *(const v8h*)(base);
    v8h hg = *(const v8h*)(base + 16);
    return __builtin_shufflevector(lo, hg, 0, 1, 2, 3, 4, 5, 6, 7,
                                           8, 9, 10, 11, 12, 13, 14, 15);
}

__global__ __launch_bounds__(256)
void StatefulMemoryModule_56367150793444_kernel(
        const float* __restrict__ query,   // [B,1,D]
        const float* __restrict__ memory,  // [B,M,D]
        const float* __restrict__ Wz, const float* __restrict__ bz,
        const float* __restrict__ Wr, const float* __restrict__ br,
        const float* __restrict__ Wc, const float* __restrict__ bc,
        float* __restrict__ out)           // [B*D read_out][B*M*D new_memory]
{
    __shared__ __align__(16) _Float16 ldsA[16 * D_DIM];   // g tile, f16, row-major
    __shared__ __align__(16) _Float16 ldsC[16 * D_DIM];   // cand tile, f16, row-major
    __shared__ __align__(16) float    ldsMem[16 * D_DIM]; // memory tile, f32
    __shared__ __align__(16) float    ldsQ[D_DIM];        // query row, f32

    const int b    = blockIdx.x;
    const int tid  = threadIdx.x;
    const int wave = tid >> 5;          // 0..7 -> owns N columns [16w,16w+16)
    const int lane = tid & 31;
    const int hi   = lane >> 4;         // lane half (A/B/C layout selector)
    const int lcol = lane & 15;
    const int col  = wave * 16 + lcol;  // this lane's output column (C layout)

    // read_out is all zeros; each block zeroes its own query row slot.
    if (tid < D_DIM) {
        out[(size_t)b * D_DIM + tid] = 0.0f;
        ldsQ[tid] = query[(size_t)b * D_DIM + tid];
    }

    // Build-phase addressing: thread handles row brow, 8 consecutive cols.
    const int brow = tid >> 4;          // 0..15
    const int bcol = (tid & 15) << 3;   // 0,8,...,120

    // ---- Load weight B-fragments once (held in registers whole kernel). ----
    // CDNA5 dense 16-bit B (32x16): lanes 0-15 cols, elems 0..7 -> K = 8*hi + 0..7,
    // elems 8..15 -> K = 16 + 8*hi + 0..7 (matches A fragment order).
    v16h wB[3][4];
    {
        const float* Ws[3] = {Wz, Wr, Wc};
        #pragma unroll
        for (int m = 0; m < 3; ++m) {
            const float* W = Ws[m];
            #pragma unroll
            for (int kf = 0; kf < 4; ++kf) {
                const int k0 = kf * 32 + hi * 8;
                v16h f;
                #pragma unroll
                for (int e = 0; e < 16; ++e)
                    f[e] = (_Float16)W[(size_t)(k0 + e + (e >= 8 ? 8 : 0)) * D_DIM + col];
                wB[m][kf] = f;
            }
        }
    }
    const float bzv = bz[col];
    const float brv = br[col];
    const float bcv = bc[col];

    __syncthreads();  // ldsQ visible

    float qreg[8];
    #pragma unroll
    for (int i = 0; i < 8; ++i) qreg[i] = ldsQ[bcol + i];
    const float qcol = ldsQ[col];

    const float* memBase = memory + (size_t)b * M_DIM * D_DIM;
    float* outMem = out + (size_t)B_DIM * D_DIM + (size_t)b * M_DIM * D_DIM;

    // ---- Software pipeline: preload tile 0 into registers. ----
    float mv[8];
    {
        const float* src = memBase + (size_t)brow * D_DIM + bcol;
        #pragma unroll
        for (int i = 0; i < 8; ++i) mv[i] = src[i];
    }

    for (int mt = 0; mt < NTILES; ++mt) {
        const int m0 = mt * 16;

        // ---- Build: mem tile (f32) and g = q + mem (f16 A) from registers. ----
        #pragma unroll
        for (int i = 0; i < 8; ++i) {
            ldsMem[brow * D_DIM + bcol + i] = mv[i];
            ldsA  [brow * D_DIM + bcol + i] = (_Float16)(mv[i] + qreg[i]);
        }
        __syncthreads();

        // ---- Prefetch next tile: loads in flight under the WMMA work. ----
        float nv[8] = {0, 0, 0, 0, 0, 0, 0, 0};
        if (mt + 1 < NTILES) {
            const float* src = memBase + (size_t)(m0 + 16 + brow) * D_DIM + bcol;
            #pragma unroll
            for (int i = 0; i < 8; ++i) nv[i] = src[i];
        }

        // ---- A fragments for g, then z and r GEMMs (bias in C init). ----
        v16h ag[4];
        #pragma unroll
        for (int kf = 0; kf < 4; ++kf) ag[kf] = load_Afrag(ldsA, lcol, kf * 32, hi);

        v8f zacc, racc;
        #pragma unroll
        for (int i = 0; i < 8; ++i) { zacc[i] = bzv; racc[i] = brv; }
        #pragma unroll
        for (int kf = 0; kf < 4; ++kf)
            zacc = __builtin_amdgcn_wmma_f32_16x16x32_f16(
                false, ag[kf], false, wB[0][kf], (short)0, zacc, false, false);
        #pragma unroll
        for (int kf = 0; kf < 4; ++kf)
            racc = __builtin_amdgcn_wmma_f32_16x16x32_f16(
                false, ag[kf], false, wB[1][kf], (short)0, racc, false, false);

        // C/D layout: acc[i], lane -> row m = i + hi*8, col = col.
        float z8[8], mem8[8];
        #pragma unroll
        for (int i = 0; i < 8; ++i) z8[i] = fast_sigmoid(zacc[i]);
        #pragma unroll
        for (int i = 0; i < 8; ++i) mem8[i] = ldsMem[(i + hi * 8) * D_DIM + col];

        // cand_in = q + r * mem  -> f16 A tile in ldsC
        #pragma unroll
        for (int i = 0; i < 8; ++i) {
            float r = fast_sigmoid(racc[i]);
            ldsC[(i + hi * 8) * D_DIM + col] = (_Float16)(qcol + r * mem8[i]);
        }
        __syncthreads();

        // ---- c = tanh(cand @ Wc + bc) ----
        v8f cacc;
        #pragma unroll
        for (int i = 0; i < 8; ++i) cacc[i] = bcv;
        #pragma unroll
        for (int kf = 0; kf < 4; ++kf) {
            v16h ac = load_Afrag(ldsC, lcol, kf * 32, hi);
            cacc = __builtin_amdgcn_wmma_f32_16x16x32_f16(
                false, ac, false, wB[2][kf], (short)0, cacc, false, false);
        }

        // ---- new_memory = (1-z)*mem + z*c ; NT stores (streamed, no reuse) ----
        #pragma unroll
        for (int i = 0; i < 8; ++i) {
            float c = fast_tanh(cacc[i]);
            float nm = mem8[i] + z8[i] * (c - mem8[i]);
            __builtin_nontemporal_store(
                nm, outMem + (size_t)(m0 + i + hi * 8) * D_DIM + col);
        }
        __syncthreads();  // guard ldsA/ldsMem/ldsC reuse next tile

        #pragma unroll
        for (int i = 0; i < 8; ++i) mv[i] = nv[i];
    }
}

extern "C" void kernel_launch(void* const* d_in, const int* in_sizes, int n_in,
                              void* d_out, int out_size, void* d_ws, size_t ws_size,
                              hipStream_t stream) {
    const float* query  = (const float*)d_in[0];
    const float* memory = (const float*)d_in[1];
    const float* Wz     = (const float*)d_in[2];
    const float* bz     = (const float*)d_in[3];
    const float* Wr     = (const float*)d_in[4];
    const float* br     = (const float*)d_in[5];
    const float* Wc     = (const float*)d_in[6];
    const float* bc     = (const float*)d_in[7];
    float* out = (float*)d_out;

    StatefulMemoryModule_56367150793444_kernel<<<dim3(B_DIM), dim3(256), 0, stream>>>(
        query, memory, Wz, bz, Wr, br, Wc, bc, out);
}
